// MaskAttention_36867999269353
// MI455X (gfx1250) — matrix operand order repaired and checked by reading, not compiled, not run on hardware
//
#include <hip/hip_runtime.h>

typedef _Float16 half_t;
typedef __attribute__((ext_vector_type(16))) _Float16 v16h;
typedef __attribute__((ext_vector_type(8)))  _Float16 v8h;
typedef __attribute__((ext_vector_type(8)))  float    v8f;

#define B_  8
#define N_  1024
#define C_  768
#define H_  12
#define D_  64
#define O3_ 2304
#define BN_ 8192
#define KT_ 24   // C_ / 32

// ---- WMMA fragment loaders (wave32, 16-bit operands) -----------------------
// A-matrix 16x32 (MxK): lanes 0-15 -> rows 0-15, K in {0..15};
// lanes 16-31 -> same rows, K in {16..31} (contiguous, matches load_b16).
__device__ __forceinline__ v16h load_a16(const half_t* base, int ld) {
  int lane = threadIdx.x & 31;
  int m    = lane & 15;
  int koff = (lane >> 4) * 16;
  const half_t* p = base + m * ld + koff;
  v8h lo = *(const v8h*)p;          // K = koff .. koff+7
  v8h hi = *(const v8h*)(p + 8);    // K = koff+8 .. koff+15
  v16h r;
  #pragma unroll
  for (int i = 0; i < 8; ++i) { r[i] = lo[i]; r[i + 8] = hi[i]; }
  return r;
}

// B-matrix 32x16 (KxN) read from a row-major tile R[n][k] (ld = row stride):
// lane = column n (lane&15), K = (lane>>4)*16 + e, e = 0..15 contiguous.
__device__ __forceinline__ v16h load_b16(const half_t* base, int ld) {
  int lane  = threadIdx.x & 31;
  int n     = lane & 15;
  int kbase = (lane >> 4) * 16;
  return *(const v16h*)(base + n * ld + kbase);  // 32B contiguous per lane
}

// ---- prep kernels ----------------------------------------------------------
__global__ void f32_to_f16_kernel(const float* __restrict__ src,
                                  half_t* __restrict__ dst, int n) {
  int i = blockIdx.x * blockDim.x + threadIdx.x;
  if (i < n) dst[i] = (half_t)src[i];
}

__global__ void mask_weight_kernel(const float* __restrict__ w,
                                   const float* __restrict__ mask,
                                   half_t* __restrict__ dst, int n) {
  int i = blockIdx.x * blockDim.x + threadIdx.x;
  if (i < n) dst[i] = (mask[i] >= 0.005f) ? (half_t)w[i] : (half_t)0.0f;
}

// ---- QKV GEMM: [BN,C] x [3C,C]^T -> q/k/vT heads ---------------------------
// Register-blocked: each wave computes a 32x64 output block
// (2 row-tiles x 4 col-tiles, 8 v8f accumulators; 8 WMMAs per K-step).
__global__ void qkv_gemm_kernel(const half_t* __restrict__ x16,
                                const half_t* __restrict__ w16,
                                const float*  __restrict__ bias,
                                half_t* __restrict__ q,
                                half_t* __restrict__ kk,
                                half_t* __restrict__ vT) {
  int wid = blockIdx.x * (blockDim.x >> 5) + (threadIdx.x >> 5);
  int mt = wid & 255;   // 256 row groups of 32 rows (BN=8192)
  int og = wid >> 8;    // 36 col groups of 64 cols (O=2304)
  v8f acc[2][4] = {};
  const half_t* xrow = x16 + (size_t)mt * 32 * C_;
  const half_t* wrow = w16 + (size_t)og * 64 * C_;
  #pragma unroll
  for (int kt = 0; kt < KT_; ++kt) {
    v16h a0 = load_a16(xrow + kt * 32,            C_);
    v16h a1 = load_a16(xrow + 16 * C_ + kt * 32,  C_);
    #pragma unroll
    for (int ct = 0; ct < 4; ++ct) {
      v16h b = load_b16(wrow + (size_t)ct * 16 * C_ + kt * 32, C_);
      acc[0][ct] = __builtin_amdgcn_wmma_f32_16x16x32_f16(false, a0, false, b,
                                            (short)0, acc[0][ct], false, false);
      acc[1][ct] = __builtin_amdgcn_wmma_f32_16x16x32_f16(false, a1, false, b,
                                            (short)0, acc[1][ct], false, false);
    }
  }
  int lane  = threadIdx.x & 31;
  int obase = og * 64;
  int part  = obase / C_;        // uniform per wave (768 % 64 == 0)
  int c2    = obase % C_;
  int h     = c2 / D_;           // uniform per wave (64-aligned groups)
  float* dummy;
  (void)dummy;
  #pragma unroll
  for (int ct = 0; ct < 4; ++ct) {
    int o  = obase + ct * 16 + (lane & 15);
    int d  = (ct * 16 + (lane & 15)) & (D_ - 1);
    float bo = bias[o];
    #pragma unroll
    for (int rt = 0; rt < 2; ++rt) {
      #pragma unroll
      for (int r = 0; r < 8; ++r) {
        int row = mt * 32 + rt * 16 + r + ((lane >> 4) << 3);
        int bb  = row >> 10;
        int n   = row & 1023;
        half_t hv = (half_t)(acc[rt][ct][r] + bo);
        size_t hb = (size_t)(bb * H_ + h) * (N_ * D_);
        if (part == 0)      q [hb + (size_t)n * D_ + d] = hv;
        else if (part == 1) kk[hb + (size_t)n * D_ + d] = hv;
        else                vT[hb + (size_t)d * N_ + n] = hv;  // V transposed
      }
    }
  }
}

// ---- flash attention: 1 wave = (b,h) x 16 query rows -----------------------
__global__ void attn_kernel(const half_t* __restrict__ q,
                            const half_t* __restrict__ k,
                            const half_t* __restrict__ vT,
                            half_t* __restrict__ ao) {
  __shared__ __align__(32) half_t lds_p[4][16 * 32];
  int wslot = threadIdx.x >> 5;
  int wid   = blockIdx.x * 4 + wslot;
  int nt = wid & 63;     // query tile 0..63
  int bh = wid >> 6;     // 0..95
  const half_t* qh = q  + (size_t)bh * (N_ * D_);
  const half_t* kh = k  + (size_t)bh * (N_ * D_);
  const half_t* vh = vT + (size_t)bh * (N_ * D_);
  half_t* myp = lds_p[wslot];
  int lane  = threadIdx.x & 31;
  int hhalf = lane >> 4;

  v16h qa0 = load_a16(qh + nt * 16 * D_ + 0,  D_);
  v16h qa1 = load_a16(qh + nt * 16 * D_ + 32, D_);

  v8f o0 = {}, o1 = {}, o2 = {}, o3 = {};
  float rm[8], rl[8];
  #pragma unroll
  for (int r = 0; r < 8; ++r) { rm[r] = -1e30f; rl[r] = 0.0f; }
  const float scale = 0.125f;  // D^-0.5

  for (int kb = 0; kb < 32; ++kb) {   // 32 key-blocks of 32 keys
    v8f s0 = {}, s1 = {};
    const half_t* kb0 = kh + (size_t)(kb * 32) * D_;
    const half_t* kb1 = kb0 + 16 * D_;
    s0 = __builtin_amdgcn_wmma_f32_16x16x32_f16(false, qa0, false, load_b16(kb0,      D_), (short)0, s0, false, false);
    s0 = __builtin_amdgcn_wmma_f32_16x16x32_f16(false, qa1, false, load_b16(kb0 + 32, D_), (short)0, s0, false, false);
    s1 = __builtin_amdgcn_wmma_f32_16x16x32_f16(false, qa0, false, load_b16(kb1,      D_), (short)0, s1, false, false);
    s1 = __builtin_amdgcn_wmma_f32_16x16x32_f16(false, qa1, false, load_b16(kb1 + 32, D_), (short)0, s1, false, false);

    float alpha[8];
    #pragma unroll
    for (int r = 0; r < 8; ++r) {
      float a = s0[r] * scale;
      float b = s1[r] * scale;
      float mx = fmaxf(a, b);
      #pragma unroll
      for (int m = 1; m < 16; m <<= 1) mx = fmaxf(mx, __shfl_xor(mx, m, 32));
      float nm = fmaxf(rm[r], mx);
      float al = __expf(rm[r] - nm);
      float p0 = __expf(a - nm);
      float p1 = __expf(b - nm);
      float ps = p0 + p1;
      #pragma unroll
      for (int m = 1; m < 16; m <<= 1) ps += __shfl_xor(ps, m, 32);
      rl[r] = rl[r] * al + ps;
      rm[r] = nm;
      alpha[r] = al;
      int row = r + (hhalf << 3);
      myp[row * 32 + (lane & 15)]      = (half_t)p0;
      myp[row * 32 + (lane & 15) + 16] = (half_t)p1;
    }
    #pragma unroll
    for (int r = 0; r < 8; ++r) {
      o0[r] *= alpha[r]; o1[r] *= alpha[r]; o2[r] *= alpha[r]; o3[r] *= alpha[r];
    }
    __syncthreads();                      // LDS cross-lane visibility (uniform)
    v16h pa = load_a16(myp, 32);          // P tile in A layout
    const half_t* vb = vh + kb * 32;
    o0 = __builtin_amdgcn_wmma_f32_16x16x32_f16(false, pa, false, load_b16(vb + 0  * N_, N_), (short)0, o0, false, false);
    o1 = __builtin_amdgcn_wmma_f32_16x16x32_f16(false, pa, false, load_b16(vb + 16 * N_, N_), (short)0, o1, false, false);
    o2 = __builtin_amdgcn_wmma_f32_16x16x32_f16(false, pa, false, load_b16(vb + 32 * N_, N_), (short)0, o2, false, false);
    o3 = __builtin_amdgcn_wmma_f32_16x16x32_f16(false, pa, false, load_b16(vb + 48 * N_, N_), (short)0, o3, false, false);
    __syncthreads();
  }

  int b_ = bh / H_;
  int h_ = bh % H_;
  #pragma unroll
  for (int r = 0; r < 8; ++r) {
    int row = nt * 16 + r + (hhalf << 3);
    float inv = 1.0f / rl[r];
    size_t base = (size_t)(b_ * N_ + row) * C_ + h_ * D_ + (lane & 15);
    ao[base + 0]  = (half_t)(o0[r] * inv);
    ao[base + 16] = (half_t)(o1[r] * inv);
    ao[base + 32] = (half_t)(o2[r] * inv);
    ao[base + 48] = (half_t)(o3[r] * inv);
  }
}

// ---- output projection: [BN,C] x [C,C]^T + bias -> fp32 --------------------
// Same 32x64 register blocking as the QKV GEMM.
__global__ void proj_gemm_kernel(const half_t* __restrict__ ao,
                                 const half_t* __restrict__ w16,
                                 const float*  __restrict__ bias,
                                 float* __restrict__ out) {
  int wid = blockIdx.x * (blockDim.x >> 5) + (threadIdx.x >> 5);
  int mt = wid & 255;   // 256 row groups of 32
  int og = wid >> 8;    // 12 col groups of 64
  v8f acc[2][4] = {};
  const half_t* arow = ao  + (size_t)mt * 32 * C_;
  const half_t* wrow = w16 + (size_t)og * 64 * C_;
  #pragma unroll
  for (int kt = 0; kt < KT_; ++kt) {
    v16h a0 = load_a16(arow + kt * 32,           C_);
    v16h a1 = load_a16(arow + 16 * C_ + kt * 32, C_);
    #pragma unroll
    for (int ct = 0; ct < 4; ++ct) {
      v16h b = load_b16(wrow + (size_t)ct * 16 * C_ + kt * 32, C_);
      acc[0][ct] = __builtin_amdgcn_wmma_f32_16x16x32_f16(false, a0, false, b,
                                            (short)0, acc[0][ct], false, false);
      acc[1][ct] = __builtin_amdgcn_wmma_f32_16x16x32_f16(false, a1, false, b,
                                            (short)0, acc[1][ct], false, false);
    }
  }
  int lane = threadIdx.x & 31;
  #pragma unroll
  for (int ct = 0; ct < 4; ++ct) {
    int col  = og * 64 + ct * 16 + (lane & 15);
    float bo = bias[col];
    #pragma unroll
    for (int rt = 0; rt < 2; ++rt) {
      #pragma unroll
      for (int r = 0; r < 8; ++r) {
        int row = mt * 32 + rt * 16 + r + ((lane >> 4) << 3);
        out[(size_t)row * C_ + col] = acc[rt][ct][r] + bo;
      }
    }
  }
}

// ---- launcher --------------------------------------------------------------
extern "C" void kernel_launch(void* const* d_in, const int* in_sizes, int n_in,
                              void* d_out, int out_size, void* d_ws, size_t ws_size,
                              hipStream_t stream) {
  const float* x     = (const float*)d_in[0];
  const float* qkvw  = (const float*)d_in[1];
  const float* qkvb  = (const float*)d_in[2];
  const float* projw = (const float*)d_in[3];
  const float* projb = (const float*)d_in[4];
  const float* maskq = (const float*)d_in[5];
  const float* maskp = (const float*)d_in[6];
  float* out = (float*)d_out;

  char* ws = (char*)d_ws;
  // byte offsets (256-aligned)
  half_t* x16  = (half_t*)(ws + 0);          // 12,582,912 B
  half_t* qw16 = (half_t*)(ws + 12582912);   //  3,538,944 B
  half_t* pw16 = (half_t*)(ws + 16121856);   //  1,179,648 B
  half_t* q    = (half_t*)(ws + 17301504);   // 12,582,912 B
  half_t* kk   = (half_t*)(ws + 29884416);   // 12,582,912 B
  half_t* vT   = (half_t*)(ws + 42467328);   // 12,582,912 B
  half_t* ao   = (half_t*)(ws + 55050240);   // 12,582,912 B  (total ~67.6 MB)

  int nx = BN_ * C_;
  f32_to_f16_kernel<<<(nx + 255) / 256, 256, 0, stream>>>(x, x16, nx);
  int nqw = O3_ * C_;
  mask_weight_kernel<<<(nqw + 255) / 256, 256, 0, stream>>>(qkvw, maskq, qw16, nqw);
  int npw = C_ * C_;
  mask_weight_kernel<<<(npw + 255) / 256, 256, 0, stream>>>(projw, maskp, pw16, npw);

  // QKV: 256 row-groups x 36 col-groups = 9216 waves, 8 waves / 256-thr block
  qkv_gemm_kernel<<<1152, 256, 0, stream>>>(x16, qw16, qkvb, q, kk, vT);
  // attention: 96 (b,h) x 64 query tiles = 6144 waves, 4 waves / 128-thr block
  attn_kernel<<<1536, 128, 0, stream>>>(q, kk, vT, ao);
  // proj: 256 row-groups x 12 col-groups = 3072 waves, 8 waves / block
  proj_gemm_kernel<<<384, 256, 0, stream>>>(ao, pw16, projb, out);
}
